// CausalSelfAttention_85366769975922
// MI455X (gfx1250) — matrix operand order repaired and checked by reading, not compiled, not run on hardware
//
#include <hip/hip_runtime.h>
#include <hip/hip_bf16.h>
#include <stdint.h>

#define B_  4
#define T_  2048
#define C_  2048
#define NH_ 16
#define HD_ 128
#define F_  (3 * C_)

typedef __attribute__((ext_vector_type(16))) __bf16 v16bf;
typedef __attribute__((ext_vector_type(8)))  float  v8f;

#define LOG2E     1.4426950408889634f
#define INV_2PI   0.15915494309189535f

__device__ __forceinline__ float fast_exp(float x)  { return __builtin_amdgcn_exp2f(x * LOG2E); }
__device__ __forceinline__ float fast_sin(float x)  { return __builtin_amdgcn_sinf(x * INV_2PI); }
__device__ __forceinline__ float fast_cos(float x)  { return __builtin_amdgcn_cosf(x * INV_2PI); }
__device__ __forceinline__ float fast_rcp(float x)  { return __builtin_amdgcn_rcpf(x); }

__device__ __forceinline__ unsigned short f2bfbits(float f) {
    unsigned u = __builtin_bit_cast(unsigned, f);
    return (unsigned short)((u + 0x7fffu + ((u >> 16) & 1u)) >> 16);  // RNE
}
__device__ __forceinline__ __bf16 f2bf(float f) {
    unsigned short s = f2bfbits(f);
    return __builtin_bit_cast(__bf16, s);
}

union Frag16 { v16bf v; uint4 q[2]; };

// Load a 32-byte WMMA fragment from two 16B-aligned chunks.
__device__ __forceinline__ v16bf ldfrag(const __bf16* p0, const __bf16* p1) {
    Frag16 u;
    u.q[0] = *(const uint4*)p0;
    u.q[1] = *(const uint4*)p1;
    return u.v;
}

__device__ __forceinline__ v8f wmma_bf16(v16bf a, v16bf b, v8f c) {
    return __builtin_amdgcn_wmma_f32_16x16x32_bf16(false, a, false, b, (short)0, c,
                                                   false, false);
}

__device__ __forceinline__ v8f zero8() {
    v8f z;
#pragma unroll
    for (int i = 0; i < 8; ++i) z[i] = 0.0f;
    return z;
}

// ---------------------------------------------------------------------------
// Kernel 1: f32 -> bf16 conversion (4 elements / thread)
// ---------------------------------------------------------------------------
__global__ __launch_bounds__(256) void f2bf_kernel(const float* __restrict__ in,
                                                   __bf16* __restrict__ out, long n) {
    long i = ((long)blockIdx.x * 256 + threadIdx.x) * 4;
    if (i + 3 < n) {
        float4 f = *(const float4*)(in + i);
        union { uint2 u; unsigned short h[4]; } pk;
        pk.h[0] = f2bfbits(f.x); pk.h[1] = f2bfbits(f.y);
        pk.h[2] = f2bfbits(f.z); pk.h[3] = f2bfbits(f.w);
        *(uint2*)(out + i) = pk.u;
    }
}

// ---------------------------------------------------------------------------
// Shared tiled-GEMM main loop: C[128x128] += A[128xK] * B[Nrows=128,K]^T
// Block = 256 threads = 8 waves in a 2(M) x 4(N) grid; each wave owns 64x32.
// LDS tiles are row-major, padded to 40 elements (80B rows, 16B aligned).
// ---------------------------------------------------------------------------
__device__ __forceinline__ void gemm_mainloop(const __bf16* __restrict__ A,
                                              const __bf16* __restrict__ Bm, int K,
                                              int bm, int bn, __bf16* As, __bf16* Bs,
                                              v8f acc[4][2]) {
    const int tid = threadIdx.x, lane = tid & 31, wave = tid >> 5;
    const int wm = wave >> 2, wn = wave & 3;
    const int nn = lane & 15, hi = lane >> 4;
    const int r = tid >> 1, half = tid & 1;  // staging: each thread copies 16 bf16
    const __bf16* ga = A  + (size_t)(bm + r) * K + half * 16;
    const __bf16* gb = Bm + (size_t)(bn + r) * K + half * 16;
    __bf16* sa = As + r * 40 + half * 16;
    __bf16* sb = Bs + r * 40 + half * 16;

    for (int kk = 0; kk < K; kk += 32) {
        *(uint4*)sa       = *(const uint4*)ga;
        *(uint4*)(sa + 8) = *(const uint4*)(ga + 8);
        *(uint4*)sb       = *(const uint4*)gb;
        *(uint4*)(sb + 8) = *(const uint4*)(gb + 8);
        ga += 32; gb += 32;
        __syncthreads();

        v16bf af[4], bfm[2];
#pragma unroll
        for (int i = 0; i < 4; ++i) {  // A frag: 16 contiguous K at hi*16
            const __bf16* p = As + (wm * 64 + i * 16 + nn) * 40 + hi * 16;
            af[i] = ldfrag(p, p + 8);
        }
#pragma unroll
        for (int j = 0; j < 2; ++j) {  // B frag: 16 contiguous K at hi*16
            const __bf16* p = Bs + (wn * 32 + j * 16 + nn) * 40 + hi * 16;
            bfm[j] = ldfrag(p, p + 8);
        }
#pragma unroll
        for (int i = 0; i < 4; ++i)
#pragma unroll
            for (int j = 0; j < 2; ++j) acc[i][j] = wmma_bf16(af[i], bfm[j], acc[i][j]);
        __syncthreads();
    }
}

// ---------------------------------------------------------------------------
// Kernel 2: qkv = x @ W_attn^T, then RoPE(q,k) (+1/sqrt(D) folded into q).
// Writes Q,K as [B,H,T,D] bf16 and V transposed as [B,H,D,T] bf16.
// ---------------------------------------------------------------------------
__global__ __launch_bounds__(256) void qkv_rope_kernel(const __bf16* __restrict__ Xb,
                                                       const __bf16* __restrict__ Wa,
                                                       __bf16* __restrict__ Qb,
                                                       __bf16* __restrict__ Kb,
                                                       __bf16* __restrict__ Vt) {
    __shared__ __align__(16) __bf16 As[128 * 40];
    __shared__ __align__(16) __bf16 Bs[128 * 40];
    v8f acc[4][2];
#pragma unroll
    for (int i = 0; i < 4; ++i)
#pragma unroll
        for (int j = 0; j < 2; ++j) acc[i][j] = zero8();

    const int bm = blockIdx.y * 128, bn = blockIdx.x * 128;
    gemm_mainloop(Xb, Wa, C_, bm, bn, As, Bs, acc);

    const int lane = threadIdx.x & 31, wave = threadIdx.x >> 5;
    const int wm = wave >> 2, wn = wave & 3, nn = lane & 15, hi = lane >> 4;
    const float kL2 = 13.287712379549449f / 64.0f;  // log2(10000)/64

#pragma unroll
    for (int j = 0; j < 2; ++j) {
        const int f  = bn + wn * 32 + j * 16 + nn;   // qkv feature index (tile-uniform region)
        const int rg = f >> 11;                      // 0=q, 1=k, 2=v
        const int cf = f & 2047;
        const int h = cf >> 7, d = cf & 127;
#pragma unroll
        for (int i = 0; i < 4; ++i) {
            const int mbase = bm + wm * 64 + i * 16;
            const int b  = mbase >> 11;              // T_ == 2048
            const int t0 = (mbase & 2047) + 8 * hi;  // this lane's 8 consecutive tokens
            if (rg == 2) {
                // V: pack 8 consecutive-t bf16 -> one b128 store into [B,H,D,T]
                union { uint4 q; unsigned short hh[8]; } pk;
#pragma unroll
                for (int rr = 0; rr < 8; ++rr) pk.hh[rr] = f2bfbits(acc[i][j][rr]);
                *(uint4*)(Vt + (((size_t)b * NH_ + h) * HD_ + d) * T_ + t0) = pk.q;
            } else {
                const int li = d >> 1;
                const float inv = __builtin_amdgcn_exp2f(-(float)li * kL2);  // 10000^(-2li/128)
                __bf16* dst = (rg == 0 ? Qb : Kb) + (((size_t)b * NH_ + h) * T_) * HD_ + d;
                const float sc = (rg == 0) ? 0.08838834764831845f : 1.0f;    // 1/sqrt(128)
#pragma unroll
                for (int rr = 0; rr < 8; ++rr) {
                    float val  = acc[i][j][rr];
                    float part = __shfl_xor(val, 1, 32);  // paired channel (d^1)
                    float ang = (float)(t0 + rr) * inv;
                    float sn = fast_sin(ang), cs = fast_cos(ang);
                    float o = (d & 1) ? (part * sn + val * cs) : (val * cs - part * sn);
                    dst[(size_t)(t0 + rr) * HD_] = f2bf(o * sc);
                }
            }
        }
    }
}

// ---------------------------------------------------------------------------
// Kernel 3: causal flash attention. Block = 4 waves; wave owns 16 q-rows.
// S = Q K^T via WMMA, online softmax (16-lane shfl reductions), P re-layout
// C->A through per-wave-private LDS (intra-wave in-order DS, no barrier),
// Y += P V via WMMA (V pre-transposed -> contiguous B-operand loads).
// ---------------------------------------------------------------------------
__global__ __launch_bounds__(128) void attn_kernel(const __bf16* __restrict__ Qb,
                                                   const __bf16* __restrict__ Kb,
                                                   const __bf16* __restrict__ Vt,
                                                   __bf16* __restrict__ Yb) {
    __shared__ __align__(16) __bf16 Pl[4][16 * 40];
    const int lane = threadIdx.x & 31, wave = threadIdx.x >> 5;
    const int nn = lane & 15, hi = lane >> 4;
    const int bh = blockIdx.y;
    const int q0 = blockIdx.x * 64 + wave * 16;

    const __bf16* Qp = Qb + (size_t)bh * T_ * HD_;
    const __bf16* Kp = Kb + (size_t)bh * T_ * HD_;
    const __bf16* Vp = Vt + (size_t)bh * HD_ * T_;

    v16bf qf[4];
#pragma unroll
    for (int kf = 0; kf < 4; ++kf) {  // A frag of Q: row = q0+nn, D chunk kf*32
        const __bf16* p = Qp + (size_t)(q0 + nn) * HD_ + kf * 32 + hi * 16;
        qf[kf] = ldfrag(p, p + 8);
    }

    v8f y[8];
#pragma unroll
    for (int ns = 0; ns < 8; ++ns) y[ns] = zero8();
    float rowmax[8], rowsum[8];
#pragma unroll
    for (int rr = 0; rr < 8; ++rr) { rowmax[rr] = -3.0e38f; rowsum[rr] = 0.0f; }

    __bf16* Pw = Pl[wave];
    const int nch = ((q0 + 15) >> 5) + 1;  // causal: k chunks of 32 up to max q row

    for (int kc = 0; kc < nch; ++kc) {
        const int k0 = kc * 32;
        v8f s0 = zero8(), s1 = zero8();
#pragma unroll
        for (int kf = 0; kf < 4; ++kf) {  // B frags of K^T: row = token, contiguous D
            const __bf16* p0 = Kp + (size_t)(k0 + nn) * HD_ + kf * 32 + hi * 16;
            const __bf16* p1 = Kp + (size_t)(k0 + 16 + nn) * HD_ + kf * 32 + hi * 16;
            v16bf kb0 = ldfrag(p0, p0 + 8);
            v16bf kb1 = ldfrag(p1, p1 + 8);
            s0 = wmma_bf16(qf[kf], kb0, s0);
            s1 = wmma_bf16(qf[kf], kb1, s1);
        }
#pragma unroll
        for (int rr = 0; rr < 8; ++rr) {
            const int qi = q0 + rr + 8 * hi;
            float a = s0[rr]; if (k0 + nn      > qi) a = -3.0e38f;
            float b = s1[rr]; if (k0 + 16 + nn > qi) b = -3.0e38f;
            float mx = fmaxf(a, b);
#pragma unroll
            for (int off = 8; off >= 1; off >>= 1) mx = fmaxf(mx, __shfl_xor(mx, off, 16));
            const float mnew = fmaxf(rowmax[rr], mx);
            const float corr = fast_exp(rowmax[rr] - mnew);   // v_exp_f32
            rowmax[rr] = mnew;
            const float pa = fast_exp(a - mnew);
            const float pb = fast_exp(b - mnew);
            float rs = pa + pb;
#pragma unroll
            for (int off = 8; off >= 1; off >>= 1) rs += __shfl_xor(rs, off, 16);
            rowsum[rr] = rowsum[rr] * corr + rs;
#pragma unroll
            for (int ns = 0; ns < 8; ++ns) y[ns][rr] *= corr;
            Pw[(rr + 8 * hi) * 40 + nn]      = f2bf(pa);
            Pw[(rr + 8 * hi) * 40 + 16 + nn] = f2bf(pb);
        }
        __builtin_amdgcn_wave_barrier();  // keep DS write->read order (in-order per wave)
        const __bf16* pp = Pw + nn * 40 + hi * 16;  // A frag of P (C->A via LDS)
        v16bf pf = ldfrag(pp, pp + 8);
        __builtin_amdgcn_wave_barrier();
#pragma unroll
        for (int ns = 0; ns < 8; ++ns) {  // B frags of V from [D,T]: contiguous tokens
            const __bf16* v0 = Vp + (size_t)(ns * 16 + nn) * T_ + k0 + hi * 16;
            v16bf vf = ldfrag(v0, v0 + 8);
            y[ns] = wmma_bf16(pf, vf, y[ns]);
        }
    }

    const int b = bh >> 4, h = bh & 15;
#pragma unroll
    for (int ns = 0; ns < 8; ++ns) {
        const int c = h * HD_ + ns * 16 + nn;
#pragma unroll
        for (int rr = 0; rr < 8; ++rr) {
            const int t = q0 + rr + 8 * hi;
            Yb[((size_t)b * T_ + t) * C_ + c] = f2bf(y[ns][rr] * fast_rcp(rowsum[rr]));
        }
    }
}

// ---------------------------------------------------------------------------
// Kernel 4: out = y @ W_proj^T (f32 output)
// ---------------------------------------------------------------------------
__global__ __launch_bounds__(256) void proj_kernel(const __bf16* __restrict__ Yb,
                                                   const __bf16* __restrict__ Wp,
                                                   float* __restrict__ Out) {
    __shared__ __align__(16) __bf16 As[128 * 40];
    __shared__ __align__(16) __bf16 Bs[128 * 40];
    v8f acc[4][2];
#pragma unroll
    for (int i = 0; i < 4; ++i)
#pragma unroll
        for (int j = 0; j < 2; ++j) acc[i][j] = zero8();

    const int bm = blockIdx.y * 128, bn = blockIdx.x * 128;
    gemm_mainloop(Yb, Wp, C_, bm, bn, As, Bs, acc);

    const int lane = threadIdx.x & 31, wave = threadIdx.x >> 5;
    const int wm = wave >> 2, wn = wave & 3, nn = lane & 15, hi = lane >> 4;
#pragma unroll
    for (int j = 0; j < 2; ++j) {
        const int n0 = bn + wn * 32 + j * 16 + nn;
#pragma unroll
        for (int i = 0; i < 4; ++i) {
            const int mb = bm + wm * 64 + i * 16 + 8 * hi;
#pragma unroll
            for (int rr = 0; rr < 8; ++rr)
                Out[(size_t)(mb + rr) * C_ + n0] = acc[i][j][rr];
        }
    }
}

// ---------------------------------------------------------------------------
extern "C" void kernel_launch(void* const* d_in, const int* in_sizes, int n_in,
                              void* d_out, int out_size, void* d_ws, size_t ws_size,
                              hipStream_t stream) {
    (void)in_sizes; (void)n_in; (void)out_size; (void)ws_size;
    const float* x      = (const float*)d_in[0];
    const float* W_attn = (const float*)d_in[1];
    const float* W_proj = (const float*)d_in[2];
    float* out = (float*)d_out;
    char* ws = (char*)d_ws;

    const long BTC = (long)B_ * T_ * C_;  // 16,777,216
    const long WAN = (long)F_ * C_;       // 12,582,912
    const long WPN = (long)C_ * C_;       //  4,194,304

    __bf16* xb  = (__bf16*)(ws + 0);           // 32 MB
    __bf16* wab = (__bf16*)(ws + 33554432);    // 24 MB
    __bf16* wpb = (__bf16*)(ws + 58720256);    //  8 MB
    __bf16* qb  = (__bf16*)(ws + 67108864);    // 32 MB
    __bf16* kb  = (__bf16*)(ws + 100663296);   // 32 MB
    __bf16* vt  = (__bf16*)(ws + 134217728);   // 32 MB (transposed V)
    __bf16* yb  = (__bf16*)(ws + 167772160);   // 32 MB   (total 192 MB)

    f2bf_kernel<<<dim3((unsigned)(BTC / 1024)), 256, 0, stream>>>(x, xb, BTC);
    f2bf_kernel<<<dim3((unsigned)(WAN / 1024)), 256, 0, stream>>>(W_attn, wab, WAN);
    f2bf_kernel<<<dim3((unsigned)(WPN / 1024)), 256, 0, stream>>>(W_proj, wpb, WPN);

    qkv_rope_kernel<<<dim3(F_ / 128, (B_ * T_) / 128), 256, 0, stream>>>(xb, wab, qb, kb, vt);
    attn_kernel<<<dim3(T_ / 64, B_ * NH_), 128, 0, stream>>>(qb, kb, vt, yb);
    proj_kernel<<<dim3(C_ / 128, (B_ * T_) / 128), 256, 0, stream>>>(yb, wpb, out);
}
